// Attention_85177791414681
// MI455X (gfx1250) — matrix operand order repaired and checked by reading, not compiled, not run on hardware
//
#include <hip/hip_runtime.h>
#include <stdint.h>

typedef __attribute__((ext_vector_type(16))) __bf16 v16bf;
typedef __attribute__((ext_vector_type(8)))  float  v8f;

#define HDIM 32
#define EDIM 16
#define VOC  27
#define SLEN 128
#define TLEN 10

static __device__ __forceinline__ float bfbits2f(uint32_t b) {
    union { uint32_t u; float f; } c; c.u = b << 16; return c.f;
}
// fast reciprocal: single v_rcp_f32, no Newton fixup (bf16-level precision is enough)
static __device__ __forceinline__ float fastrcp(float x) {
    return __builtin_amdgcn_rcpf(x);
}
static __device__ __forceinline__ float sigmoidf_(float x) {
    return fastrcp(1.0f + __expf(-x));
}
static __device__ __forceinline__ float tanhf_(float x) {
#if __has_builtin(__builtin_amdgcn_tanhf)
    return __builtin_amdgcn_tanhf(x);
#else
    float e = __expf(2.0f * x);
    return 1.0f - 2.0f * fastrcp(e + 1.0f);
#endif
}
// A-matrix (16x32 bf16) per-lane K mapping: lanes0-15: K=0..7,16..23 ; lanes16-31: +8
static __device__ __forceinline__ int kmapA(int i, int lane) {
    return ((i & 8) ? (i + 8) : i) + ((lane & 16) ? 8 : 0);
}
// B-matrix (32x16 bf16) per-lane K mapping: lanes0-15: K=0..7,16..23 ; lanes16-31: +8
static __device__ __forceinline__ int kmapB(int i, int lane) {
    return ((i & 8) ? (i + 8) : i) + ((lane & 16) ? 8 : 0);
}
static __device__ __forceinline__ v8f wmma_bf16(v16bf a, v16bf b, v8f c) {
    return __builtin_amdgcn_wmma_f32_16x16x32_bf16(false, a, false, b, (short)0, c,
                                                   false, false);
}

// A fragment from bf16 LDS tile z[16][64], K-chunk starting at k0
static __device__ __forceinline__ v16bf afrag_z(const __bf16 (*z)[64], int lane, int k0) {
    int m = lane & 15;
    v16bf a;
    #pragma unroll
    for (int i = 0; i < 16; ++i) a[i] = z[m][k0 + kmapA(i, lane)];
    return a;
}
// A fragment from f32 LDS tile src[16][32] (single K chunk)
static __device__ __forceinline__ v16bf afrag_f32(const float (*src)[32], int lane) {
    int m = lane & 15;
    v16bf a;
    #pragma unroll
    for (int i = 0; i < 16; ++i) a[i] = (__bf16)src[m][kmapA(i, lane)];
    return a;
}
// B fragment of combined LSTM weight  B[k][n] = [Wih | Whh][n][k], K padded 48->64
static __device__ __forceinline__ v16bf bfrag_lstm(const float* Wih, const float* Whh,
                                                   int lane, int nt, int kc) {
    int n = nt * 16 + (lane & 15);
    v16bf b;
    #pragma unroll
    for (int i = 0; i < 16; ++i) {
        int k = kc * 32 + kmapB(i, lane);
        float w = 0.0f;
        if (k < EDIM)               w = Wih[n * EDIM + k];
        else if (k < EDIM + HDIM)   w = Whh[n * HDIM + (k - EDIM)];
        b[i] = (__bf16)w;
    }
    return b;
}
// B fragment of generic row-major W[nrows][ldk] (B[k][n] = W[n][k])
static __device__ __forceinline__ v16bf bfrag_rm(const float* W, int ldk, int nrows,
                                                 int lane, int nt, int kc) {
    int n = nt * 16 + (lane & 15);
    v16bf b;
    #pragma unroll
    for (int i = 0; i < 16; ++i) {
        int k = kc * 32 + kmapB(i, lane);
        float w = (n < nrows && k < ldk) ? W[n * ldk + k] : 0.0f;
        b[i] = (__bf16)w;
    }
    return b;
}

// One LSTM cell step for 16 rows: z[m][0..47] = [emb(16), h(32)], pad 48..63 = 0.
// Gates via 16 WMMAs; c kept in registers (C/D lane layout makes i/f/g/o for a
// given (row, hcol) land in the same lane across tiles 2q+hh); new h -> z[m][16..47].
static __device__ __forceinline__ void lstm_step(__bf16 (*z)[64], const v16bf wb[2][8],
                                                 const float* bias, float c_reg[2][8],
                                                 int lane) {
    v16bf a0 = afrag_z((const __bf16 (*)[64])z, lane, 0);
    v16bf a1 = afrag_z((const __bf16 (*)[64])z, lane, 32);
    const int n_lane = lane & 15;
    v8f acc[8];
    #pragma unroll
    for (int nt = 0; nt < 8; ++nt) {
        float bv = bias[nt * 16 + n_lane];
        v8f c;
        #pragma unroll
        for (int j = 0; j < 8; ++j) c[j] = bv;
        c = wmma_bf16(a0, wb[0][nt], c);
        c = wmma_bf16(a1, wb[1][nt], c);
        acc[nt] = c;
    }
    #pragma unroll
    for (int hh = 0; hh < 2; ++hh) {
        #pragma unroll
        for (int j = 0; j < 8; ++j) {
            float ig = sigmoidf_(acc[0 + hh][j]);
            float fg = sigmoidf_(acc[2 + hh][j]);
            float gg = tanhf_  (acc[4 + hh][j]);
            float og = sigmoidf_(acc[6 + hh][j]);
            float cn = fg * c_reg[hh][j] + ig * gg;
            float hn = og * tanhf_(cn);
            c_reg[hh][j] = cn;
            int m = j + ((lane >> 4) << 3);
            z[m][EDIM + hh * 16 + n_lane] = (__bf16)hn;
        }
    }
}

extern "C" __global__ __launch_bounds__(128, 1)
void seq2seq_attn_kernel(const int* __restrict__ x, const int* __restrict__ y,
                         const float* __restrict__ enc_embed, const float* __restrict__ dec_embed,
                         const float* __restrict__ enc_Wih, const float* __restrict__ enc_Whh,
                         const float* __restrict__ enc_b,
                         const float* __restrict__ dec_Wih, const float* __restrict__ dec_Whh,
                         const float* __restrict__ dec_b,
                         const float* __restrict__ attn_W, const float* __restrict__ attn_b,
                         const float* __restrict__ out_W, const float* __restrict__ out_b,
                         float* __restrict__ out, unsigned short* __restrict__ outx_raw)
{
    __shared__ __bf16 z_s[4][16][64];   //  8 KB: [emb | h | pad] staging (bf16)
    __shared__ float  a_s[4][16][32];   //  8 KB: attention query a
    __shared__ float  sc_s[4][16][128]; // 32 KB: attention scores
    __shared__ float  cx_s[4][16][32];  //  8 KB: context vector

    const int tid    = threadIdx.x;
    const int wid    = tid >> 5;
    const int lane   = tid & 31;
    const int row0   = (blockIdx.x * 4 + wid) * 16;
    const int n_lane = lane & 15;

    __bf16* const outx = (__bf16*)outx_raw;
    __bf16 (*z)[64]  = z_s[wid];
    float (*aat)[32] = a_s[wid];
    float (*sc)[128] = sc_s[wid];
    float (*ctx)[32] = cx_s[wid];

    // ---- encoder weights -> register-resident B fragments ----
    v16bf wb[2][8];
    #pragma unroll
    for (int nt = 0; nt < 8; ++nt) {
        wb[0][nt] = bfrag_lstm(enc_Wih, enc_Whh, lane, nt, 0);
        wb[1][nt] = bfrag_lstm(enc_Wih, enc_Whh, lane, nt, 1);
    }

    float c_reg[2][8];
    #pragma unroll
    for (int hh = 0; hh < 2; ++hh) {
        #pragma unroll
        for (int j = 0; j < 8; ++j) c_reg[hh][j] = 0.0f;
    }
    for (int e = lane; e < 16 * 48; e += 32) {     // h = 0 and K-pad = 0
        int m = e / 48, k = 16 + e % 48;
        z[m][k] = (__bf16)0.0f;
    }

    // ================= encoder scan (S = 128 sequential steps) =================
    for (int t = 0; t < SLEN; ++t) {
        {   // stage x_t embedding (bf16) into z[:,0..15]
            int m   = n_lane;
            int tok = x[(long)(row0 + m) * SLEN + t];
            int e0  = (lane >> 4) * 8;
            #pragma unroll
            for (int e = 0; e < 8; ++e)
                z[m][e0 + e] = (__bf16)enc_embed[tok * EDIM + e0 + e];
        }
        lstm_step(z, wb, enc_b, c_reg, lane);
        // coalesced 64B-per-row bf16 store of h -> out_x (L2 resident: 134 MB)
        #pragma unroll
        for (int r = 0; r < 16; ++r)
            outx[((long)(row0 + r) * SLEN + t) * HDIM + lane] = z[r][EDIM + lane];
    }

    // ---- decoder weights ----
    #pragma unroll
    for (int nt = 0; nt < 8; ++nt) {
        wb[0][nt] = bfrag_lstm(dec_Wih, dec_Whh, lane, nt, 0);
        wb[1][nt] = bfrag_lstm(dec_Wih, dec_Whh, lane, nt, 1);
    }
    v16bf wa[2];
    wa[0] = bfrag_rm(attn_W, HDIM, HDIM, lane, 0, 0);
    wa[1] = bfrag_rm(attn_W, HDIM, HDIM, lane, 1, 0);
    v16bf wo[2][2];
    #pragma unroll
    for (int nt = 0; nt < 2; ++nt) {
        wo[0][nt] = bfrag_rm(out_W, 2 * HDIM, VOC, lane, nt, 0);
        wo[1][nt] = bfrag_rm(out_W, 2 * HDIM, VOC, lane, nt, 1);
    }

    const __bf16* oxbase = outx + (long)row0 * SLEN * HDIM;

    // ================= decoder scan (T = 10 steps) =================
    for (int t = 0; t < TLEN; ++t) {
        // ---- a = h @ attn_W^T + attn_b (h lives in z[:,16..47]) ----
        {
            v16bf ah = afrag_z((const __bf16 (*)[64])z, lane, EDIM);
            #pragma unroll
            for (int nt = 0; nt < 2; ++nt) {
                float bv = attn_b[nt * 16 + n_lane];
                v8f c;
                #pragma unroll
                for (int j = 0; j < 8; ++j) c[j] = bv;
                c = wmma_bf16(ah, wa[nt], c);
                #pragma unroll
                for (int j = 0; j < 8; ++j) {
                    int m = j + ((lane >> 4) << 3);
                    aat[m][nt * 16 + n_lane] = c[j];
                }
            }
        }
        // ---- scores[r][s] = dot(a[r], out_x[r][s]) ; 64B vector loads from L2 ----
        #pragma unroll 1
        for (int r = 0; r < 16; ++r) {
            float areg[HDIM];
            #pragma unroll
            for (int k = 0; k < HDIM; ++k) areg[k] = aat[r][k];
            #pragma unroll
            for (int ss = 0; ss < 4; ++ss) {
                int s = ss * 32 + lane;
                const uint32_t* p = (const uint32_t*)(oxbase + ((long)r * SLEN + s) * HDIM);
                float acc = 0.0f;
                #pragma unroll
                for (int kk = 0; kk < 16; ++kk) {
                    uint32_t u = p[kk];
                    acc += areg[2 * kk]     * bfbits2f(u & 0xffffu);
                    acc += areg[2 * kk + 1] * bfbits2f(u >> 16);
                }
                sc[r][s] = acc;
            }
        }
        // ---- softmax over s=128 per row (wave32 shfl reductions) ----
        #pragma unroll 1
        for (int r = 0; r < 16; ++r) {
            float v0 = sc[r][lane],      v1 = sc[r][lane + 32],
                  v2 = sc[r][lane + 64], v3 = sc[r][lane + 96];
            float mx = fmaxf(fmaxf(v0, v1), fmaxf(v2, v3));
            #pragma unroll
            for (int off = 16; off > 0; off >>= 1) mx = fmaxf(mx, __shfl_xor(mx, off, 32));
            float e0 = __expf(v0 - mx), e1 = __expf(v1 - mx),
                  e2 = __expf(v2 - mx), e3 = __expf(v3 - mx);
            float sm = e0 + e1 + e2 + e3;
            #pragma unroll
            for (int off = 16; off > 0; off >>= 1) sm += __shfl_xor(sm, off, 32);
            float inv = fastrcp(sm);
            sc[r][lane] = e0 * inv;      sc[r][lane + 32] = e1 * inv;
            sc[r][lane + 64] = e2 * inv; sc[r][lane + 96] = e3 * inv;
        }
        // ---- ctx[r][d] = sum_s p[s] * out_x[r][s][d] (lane = d, coalesced) ----
        #pragma unroll 1
        for (int r = 0; r < 16; ++r) {
            float acc = 0.0f;
            const __bf16* pr = oxbase + (long)r * SLEN * HDIM + lane;
            #pragma unroll 4
            for (int s = 0; s < SLEN; ++s)
                acc += sc[r][s] * (float)pr[(long)s * HDIM];
            ctx[r][lane] = acc;
        }
        // ---- decoder LSTM cell on y_t embedding ----
        {
            int m   = n_lane;
            int tok = y[(long)(row0 + m) * (TLEN + 1) + t];
            int e0  = (lane >> 4) * 8;
            #pragma unroll
            for (int e = 0; e < 8; ++e)
                z[m][e0 + e] = (__bf16)dec_embed[tok * EDIM + e0 + e];
        }
        lstm_step(z, wb, dec_b, c_reg, lane);
        // ---- out = [ctx, h_new] @ out_W^T + out_b  (K = 32 + 32, N padded 27->32) ----
        {
            v16bf ac  = afrag_f32((const float (*)[32])ctx, lane);
            v16bf ahn = afrag_z((const __bf16 (*)[64])z, lane, EDIM);
            #pragma unroll
            for (int nt = 0; nt < 2; ++nt) {
                int n = nt * 16 + n_lane;
                float bv = (n < VOC) ? out_b[n] : 0.0f;
                v8f c;
                #pragma unroll
                for (int j = 0; j < 8; ++j) c[j] = bv;
                c = wmma_bf16(ac,  wo[0][nt], c);
                c = wmma_bf16(ahn, wo[1][nt], c);
                if (n < VOC) {
                    #pragma unroll
                    for (int j = 0; j < 8; ++j) {
                        int m = j + ((lane >> 4) << 3);
                        out[((long)(row0 + m) * TLEN + t) * VOC + n] = c[j];
                    }
                }
            }
        }
    }
}

extern "C" void kernel_launch(void* const* d_in, const int* in_sizes, int n_in,
                              void* d_out, int out_size, void* d_ws, size_t ws_size,
                              hipStream_t stream) {
    (void)n_in; (void)out_size; (void)ws_size;
    const int*   x         = (const int*)d_in[0];
    const int*   y         = (const int*)d_in[1];
    const float* enc_embed = (const float*)d_in[2];
    const float* dec_embed = (const float*)d_in[3];
    const float* enc_Wih   = (const float*)d_in[4];
    const float* enc_Whh   = (const float*)d_in[5];
    const float* enc_b     = (const float*)d_in[6];
    const float* dec_Wih   = (const float*)d_in[7];
    const float* dec_Whh   = (const float*)d_in[8];
    const float* dec_b     = (const float*)d_in[9];
    const float* attn_W    = (const float*)d_in[10];
    const float* attn_b    = (const float*)d_in[11];
    const float* out_W     = (const float*)d_in[12];
    const float* out_b     = (const float*)d_in[13];
    float*       out       = (float*)d_out;
    unsigned short* outx   = (unsigned short*)d_ws;  // bf16 out_x: B*S*H*2 = 134 MB

    int B    = in_sizes[0] / SLEN;   // 16384
    int grid = B / 64;               // 4 waves/WG * 16 rows/wave
    seq2seq_attn_kernel<<<grid, 128, 0, stream>>>(
        x, y, enc_embed, dec_embed, enc_Wih, enc_Whh, enc_b,
        dec_Wih, dec_Whh, dec_b, attn_W, attn_b, out_W, out_b, out, outx);
}
